// Attention_36988258353160
// MI455X (gfx1250) — matrix operand order repaired and checked by reading, not compiled, not run on hardware
//
#include <hip/hip_runtime.h>

// ---------------------------------------------------------------------------
// DeepViT re-attention for MI455X (gfx1250, wave32, WMMA).
// All matmuls use v_wmma_f32_16x16x32_bf16 (bf16 in, f32 accumulate).
// Attention is flash-style two-pass per 16-query tile with all 8 heads
// co-resident in one workgroup (required by the post-softmax head mixing
// + LayerNorm-over-heads). No [n,n] tensor ever touches global memory.
// Pipelined loops are unrolled x2 with ping-pong register sets so the
// double-buffer rotation from round 2 (v_dual_mov_b32 storms) disappears:
// loads land directly in the registers the next WMMA block consumes.
// ---------------------------------------------------------------------------

typedef __bf16 bf16;
typedef bf16  v16bf __attribute__((ext_vector_type(16)));
typedef bf16  v8bf  __attribute__((ext_vector_type(8)));
typedef float v8f   __attribute__((ext_vector_type(8)));

#define BATCH 2
#define NSEQ  2048
#define DIM   512
#define HEADS 8
#define DHEAD 64
#define INNER 512
#define MTOT  (BATCH * NSEQ)          // 4096 rows for the two dense GEMMs

__device__ __forceinline__ bf16 f2bf(float f) {
  unsigned u = __float_as_uint(f);
  unsigned r = u + 0x7FFFu + ((u >> 16) & 1u);   // round-to-nearest-even
  unsigned short h = (unsigned short)(r >> 16);
  return __builtin_bit_cast(bf16, h);
}

__device__ __forceinline__ v8f wmma_bf16(v16bf a, v16bf b, v8f c) {
  // D = A(16x32) * B(32x16) + C(16x16 f32)
  return __builtin_amdgcn_wmma_f32_16x16x32_bf16(
      /*neg_a=*/false, a, /*neg_b=*/false, b,
      /*c_mod=*/(short)0, c, /*reuse_a=*/false, /*reuse_b=*/false);
}

// A-operand load (16-bit A 16x32): row = lane&15,
// elems 0..15 <- K = k0 + 16*half + 0..15 (matches load_B16)
__device__ __forceinline__ v16bf load_A(const bf16* base, int row, int ld,
                                        int k0, int half) {
  const bf16* p = base + row * ld + k0 + half * 16;
  v8bf lo = *(const v8bf*)p;
  v8bf hi = *(const v8bf*)(p + 8);
  v16bf r;
#pragma unroll
  for (int i = 0; i < 8; ++i) { r[i] = lo[i]; r[i + 8] = hi[i]; }
  return r;
}

// B-operand load (16-bit B 32x16, contiguous k order used here): col = lane&15,
// elems 0..15 <- K = k0 + 16*half + 0..15  (one contiguous 32B load)
__device__ __forceinline__ v16bf load_B16(const bf16* p) {
  return *(const v16bf*)p;
}

// ---------------------------------------------------------------------------
// Kernel 1: fp32 -> bf16 conversion of x, w_qkv, w_out
// ---------------------------------------------------------------------------
__global__ void cvt_kernel(const float* __restrict__ x,
                           const float* __restrict__ wqkv,
                           const float* __restrict__ wout,
                           bf16* __restrict__ xb,
                           bf16* __restrict__ wqkvb,
                           bf16* __restrict__ woutb) {
  int i = blockIdx.x * blockDim.x + threadIdx.x;
  if (i < MTOT * DIM)          xb[i]    = f2bf(x[i]);
  if (i < 3 * INNER * DIM)     wqkvb[i] = f2bf(wqkv[i]);
  if (i < DIM * INNER)         woutb[i] = f2bf(wout[i]);
}

// ---------------------------------------------------------------------------
// Kernel 2: QKV GEMM  qkv[m,o] = sum_k xb[m,k] * Wqkv[o,k]
// 16(M) x 64(N) tile per wave; k-loop unrolled x2 with ping-pong operand
// sets (no rotation copies). Epilogue scatters into Q/K row-major + Vt.
// ---------------------------------------------------------------------------
__global__ __launch_bounds__(256) void qkv_gemm_kernel(
    const bf16* __restrict__ xb, const bf16* __restrict__ wb,
    bf16* __restrict__ q, bf16* __restrict__ k, bf16* __restrict__ vT) {
  const int lane = threadIdx.x & 31;
  const int wave = threadIdx.x >> 5;
  const int half = lane >> 4;
  const int lc   = lane & 15;
  const int NTN  = (3 * INNER) / 64;                 // 24 n-chunks
  int idx = blockIdx.x * 8 + wave;                   // 6144 wave-tiles
  int mt = idx / NTN, nc = idx % NTN;
  int m0 = mt * 16, o0 = nc * 64;
  const int row = m0 + lc;

  v8f acc[4] = {v8f{}, v8f{}, v8f{}, v8f{}};

  // ping-pong operand sets
  v16bf a0 = load_A(xb, row, DIM, 0, half);
  v16bf b0[4];
#pragma unroll
  for (int t = 0; t < 4; ++t)
    b0[t] = load_B16(wb + half * 16 + (o0 + t * 16 + lc) * DIM);
  v16bf a1, b1[4];

  for (int k0 = 0; k0 < DIM - 64; k0 += 64) {
    // load set1 (k0+32) ahead of consuming set0 (k0)
    a1 = load_A(xb, row, DIM, k0 + 32, half);
    const bf16* bp1 = wb + (k0 + 32) + half * 16;
#pragma unroll
    for (int t = 0; t < 4; ++t) b1[t] = load_B16(bp1 + (o0 + t * 16 + lc) * DIM);
#pragma unroll
    for (int t = 0; t < 4; ++t) acc[t] = wmma_bf16(a0, b0[t], acc[t]);
    // load set0 (k0+64) ahead of consuming set1 (k0+32)
    a0 = load_A(xb, row, DIM, k0 + 64, half);
    const bf16* bp0 = wb + (k0 + 64) + half * 16;
    __builtin_prefetch(bp0 + 64 + (o0 + lc) * DIM, 0, 1);  // global_prefetch_b8
#pragma unroll
    for (int t = 0; t < 4; ++t) b0[t] = load_B16(bp0 + (o0 + t * 16 + lc) * DIM);
#pragma unroll
    for (int t = 0; t < 4; ++t) acc[t] = wmma_bf16(a1, b1[t], acc[t]);
  }
  // tail: set0 holds k=DIM-64; load set1 at DIM-32 and drain both
  a1 = load_A(xb, row, DIM, DIM - 32, half);
  {
    const bf16* bp1 = wb + (DIM - 32) + half * 16;
#pragma unroll
    for (int t = 0; t < 4; ++t) b1[t] = load_B16(bp1 + (o0 + t * 16 + lc) * DIM);
  }
#pragma unroll
  for (int t = 0; t < 4; ++t) acc[t] = wmma_bf16(a0, b0[t], acc[t]);
#pragma unroll
  for (int t = 0; t < 4; ++t) acc[t] = wmma_bf16(a1, b1[t], acc[t]);

#pragma unroll
  for (int t = 0; t < 4; ++t) {
    int o = o0 + t * 16 + lc;
#pragma unroll
    for (int v = 0; v < 8; ++v) {
      int m  = m0 + v + 8 * half;
      int bb = m >> 11;                // m / NSEQ
      int ni = m & (NSEQ - 1);
      bf16 val = f2bf(acc[t][v]);
      if (o < INNER) {                                  // Q
        int h = o >> 6, d = o & 63;
        q[(((bb * HEADS + h) * NSEQ) + ni) * DHEAD + d] = val;
      } else if (o < 2 * INNER) {                       // K
        int oo = o - INNER; int h = oo >> 6, d = oo & 63;
        k[(((bb * HEADS + h) * NSEQ) + ni) * DHEAD + d] = val;
      } else {                                          // V, transposed [d][n]
        int oo = o - 2 * INNER; int h = oo >> 6, d = oo & 63;
        vT[((bb * HEADS + h) * DHEAD + d) * NSEQ + ni] = val;
      }
    }
  }
}

// ---------------------------------------------------------------------------
// Kernel 3: fused attention + re-attention.
// Grid: BATCH * (NSEQ/16) workgroups, 256 threads = 8 waves, wave h = head h.
// Pass 1: online softmax stats via QK^T WMMAs, ping-pong K-tile loads,
//         half-wave shuffle reduction. Pass 2: V tiles hoisted over the LDS
//         barriers; dots recomputed, probs staged in LDS, 8x8 head mix +
//         LN over heads, bf16 A-operand from LDS, P x V WMMAs.
// ---------------------------------------------------------------------------
__global__ __launch_bounds__(256) void attn_kernel(
    const bf16* __restrict__ q, const bf16* __restrict__ k,
    const bf16* __restrict__ vT,
    const float* __restrict__ reattn_w, const float* __restrict__ ln_g,
    const float* __restrict__ ln_b, bf16* __restrict__ aout) {
  __shared__ float pbuf[HEADS][16][32];   // 16 KB: per-head normalized probs
  __shared__ bf16  abuf[HEADS][16][32];   //  8 KB: mixed+LN probs (A feed)
  __shared__ float w_s[HEADS][HEADS];
  __shared__ float g_s[HEADS], b_s[HEADS];

  const int tid  = threadIdx.x;
  const int lane = tid & 31;
  const int h    = tid >> 5;              // wave id == head id
  const int half = lane >> 4;
  const int lc   = lane & 15;
  const int bi   = blockIdx.x >> 7;       // 128 query tiles per batch
  const int i0   = (blockIdx.x & 127) * 16;
  const float scale = 0.125f;             // DIM_HEAD^-0.5

  if (tid < 64) w_s[tid >> 3][tid & 7] = reattn_w[tid];
  if (tid < 8) { g_s[tid] = ln_g[tid]; b_s[tid] = ln_b[tid]; }
  __syncthreads();

  const bf16* qh = q  + (size_t)(bi * HEADS + h) * NSEQ * DHEAD;
  const bf16* kh = k  + (size_t)(bi * HEADS + h) * NSEQ * DHEAD;
  const bf16* vh = vT + (size_t)(bi * HEADS + h) * DHEAD * NSEQ;

  // Q A-operands for the two 32-wide k chunks of d=64 (live whole kernel)
  v16bf qa0 = load_A(qh, i0 + lc, DHEAD, 0, half);
  v16bf qa1 = load_A(qh, i0 + lc, DHEAD, 32, half);

  // ---- Pass 1: online softmax stats (ping-pong K tiles) ------------------
  float m[8], l[8];
#pragma unroll
  for (int v = 0; v < 8; ++v) { m[v] = -3.0e38f; l[v] = 0.0f; }

  auto dots_update = [&](v16bf kb0, v16bf kb1) {
    v8f s = v8f{};
    s = wmma_bf16(qa0, kb0, s);
    s = wmma_bf16(qa1, kb1, s);
#pragma unroll
    for (int v = 0; v < 8; ++v) {
      float sv = s[v] * scale;
      float mn = fmaxf(m[v], sv);
      l[v] = l[v] * __expf(m[v] - mn) + __expf(sv - mn);
      m[v] = mn;
    }
  };

  const bf16* kbase = kh + lc * DHEAD + half * 16;
  v16bf c0 = load_B16(kbase);
  v16bf c1 = load_B16(kbase + 32);
  v16bf d0, d1;
  for (int j0 = 0; j0 < NSEQ - 32; j0 += 32) {
    const bf16* kp1 = kh + (j0 + 16 + lc) * DHEAD + half * 16;
    d0 = load_B16(kp1);
    d1 = load_B16(kp1 + 32);
    dots_update(c0, c1);
    const bf16* kp0 = kh + (j0 + 32 + lc) * DHEAD + half * 16;
    c0 = load_B16(kp0);
    c1 = load_B16(kp0 + 32);
    dots_update(d0, d1);
  }
  {
    const bf16* kpt = kh + (NSEQ - 16 + lc) * DHEAD + half * 16;
    d0 = load_B16(kpt);
    d1 = load_B16(kpt + 32);
    dots_update(c0, c1);
    dots_update(d0, d1);
  }

  // combine the 16 per-column partials; xor masks < 16 stay in each half-wave
#pragma unroll
  for (int mask = 1; mask < 16; mask <<= 1) {
#pragma unroll
    for (int v = 0; v < 8; ++v) {
      float om = __shfl_xor(m[v], mask, 32);
      float ol = __shfl_xor(l[v], mask, 32);
      float mn = fmaxf(m[v], om);
      l[v] = l[v] * __expf(m[v] - mn) + ol * __expf(om - mn);
      m[v] = mn;
    }
  }
  float rl[8];
#pragma unroll
  for (int v = 0; v < 8; ++v) rl[v] = 1.0f / l[v];

  // ---- Pass 2: probs -> head mix -> LN -> P x V --------------------------
  v8f o[4] = {v8f{}, v8f{}, v8f{}, v8f{}};
  for (int j0 = 0; j0 < NSEQ; j0 += 32) {
    // V B-tiles: independent of the barriers below -> issue first, consume
    // after the second __syncthreads() (loads hide under dots + mixing).
    const bf16* vp = vh + j0 + half * 16;
    v16bf vb[4];
#pragma unroll
    for (int t = 0; t < 4; ++t) vb[t] = load_B16(vp + (t * 16 + lc) * NSEQ);

#pragma unroll
    for (int sub = 0; sub < 2; ++sub) {
      const bf16* kp = kh + (j0 + sub * 16 + lc) * DHEAD + half * 16;
      v16bf kb0 = load_B16(kp);
      v16bf kb1 = load_B16(kp + 32);
      v8f s = v8f{};
      s = wmma_bf16(qa0, kb0, s);
      s = wmma_bf16(qa1, kb1, s);
#pragma unroll
      for (int v = 0; v < 8; ++v)
        pbuf[h][v + 8 * half][sub * 16 + lc] =
            __expf(s[v] * scale - m[v]) * rl[v];
    }
    __syncthreads();

    // 512 (i,j) positions, 256 threads, 2 each: mix across heads + LN(heads)
#pragma unroll
    for (int rep = 0; rep < 2; ++rep) {
      int pos = tid + rep * 256;
      int ii = pos >> 5, jj = pos & 31;
      float ph[HEADS];
#pragma unroll
      for (int hh = 0; hh < HEADS; ++hh) ph[hh] = pbuf[hh][ii][jj];
      float mix[HEADS]; float mu = 0.0f;
#pragma unroll
      for (int g = 0; g < HEADS; ++g) {
        float a = 0.0f;
#pragma unroll
        for (int hh = 0; hh < HEADS; ++hh) a += ph[hh] * w_s[hh][g];
        mix[g] = a; mu += a;
      }
      mu *= 0.125f;
      float var = 0.0f;
#pragma unroll
      for (int g = 0; g < HEADS; ++g) { float d = mix[g] - mu; var += d * d; }
      var *= 0.125f;
      float rs = rsqrtf(var + 1e-5f);
#pragma unroll
      for (int g = 0; g < HEADS; ++g)
        abuf[g][ii][jj] = f2bf((mix[g] - mu) * rs * g_s[g] + b_s[g]);
    }
    __syncthreads();

    // wave h now acts as output head g=h: A = abuf[h] (16x32), B = vb[]
    v16bf pa = load_A(&abuf[h][0][0], lc, 32, 0, half);
#pragma unroll
    for (int t = 0; t < 4; ++t) o[t] = wmma_bf16(pa, vb[t], o[t]);
  }

  // epilogue: [b, n, h*64+d] bf16, ready as A-operand of the out-projection
#pragma unroll
  for (int t = 0; t < 4; ++t) {
    int d = t * 16 + lc;
#pragma unroll
    for (int v = 0; v < 8; ++v) {
      int irow = i0 + v + 8 * half;
      aout[((size_t)(bi * NSEQ + irow)) * INNER + h * DHEAD + d] = f2bf(o[t][v]);
    }
  }
}

// ---------------------------------------------------------------------------
// Kernel 4: out = aout @ w_out^T + b_out  (fp32 result); ping-pong k-loop
// ---------------------------------------------------------------------------
__global__ __launch_bounds__(256) void out_gemm_kernel(
    const bf16* __restrict__ ab, const bf16* __restrict__ wb,
    const float* __restrict__ bias, float* __restrict__ out) {
  const int lane = threadIdx.x & 31;
  const int wave = threadIdx.x >> 5;
  const int half = lane >> 4;
  const int lc   = lane & 15;
  const int NTN  = DIM / 64;                       // 8 n-chunks
  int idx = blockIdx.x * 8 + wave;                 // 2048 wave-tiles
  int mt = idx / NTN, nc = idx % NTN;
  int m0 = mt * 16, o0 = nc * 64;
  const int row = m0 + lc;

  v8f acc[4] = {v8f{}, v8f{}, v8f{}, v8f{}};

  v16bf a0 = load_A(ab, row, INNER, 0, half);
  v16bf b0[4];
#pragma unroll
  for (int t = 0; t < 4; ++t)
    b0[t] = load_B16(wb + half * 16 + (o0 + t * 16 + lc) * INNER);
  v16bf a1, b1[4];

  for (int k0 = 0; k0 < INNER - 64; k0 += 64) {
    a1 = load_A(ab, row, INNER, k0 + 32, half);
    const bf16* bp1 = wb + (k0 + 32) + half * 16;
#pragma unroll
    for (int t = 0; t < 4; ++t) b1[t] = load_B16(bp1 + (o0 + t * 16 + lc) * INNER);
#pragma unroll
    for (int t = 0; t < 4; ++t) acc[t] = wmma_bf16(a0, b0[t], acc[t]);
    a0 = load_A(ab, row, INNER, k0 + 64, half);
    const bf16* bp0 = wb + (k0 + 64) + half * 16;
#pragma unroll
    for (int t = 0; t < 4; ++t) b0[t] = load_B16(bp0 + (o0 + t * 16 + lc) * INNER);
#pragma unroll
    for (int t = 0; t < 4; ++t) acc[t] = wmma_bf16(a1, b1[t], acc[t]);
  }
  a1 = load_A(ab, row, INNER, INNER - 32, half);
  {
    const bf16* bp1 = wb + (INNER - 32) + half * 16;
#pragma unroll
    for (int t = 0; t < 4; ++t) b1[t] = load_B16(bp1 + (o0 + t * 16 + lc) * INNER);
  }
#pragma unroll
  for (int t = 0; t < 4; ++t) acc[t] = wmma_bf16(a0, b0[t], acc[t]);
#pragma unroll
  for (int t = 0; t < 4; ++t) acc[t] = wmma_bf16(a1, b1[t], acc[t]);

#pragma unroll
  for (int t = 0; t < 4; ++t) {
    int oidx = o0 + t * 16 + lc;
    float bv = bias[oidx];
#pragma unroll
    for (int v = 0; v < 8; ++v) {
      int mm = m0 + v + 8 * half;
      out[(size_t)mm * DIM + oidx] = acc[t][v] + bv;
    }
  }
}

// ---------------------------------------------------------------------------
// Host-side launcher (graph-capture safe: only kernel launches on `stream`)
// ---------------------------------------------------------------------------
extern "C" void kernel_launch(void* const* d_in, const int* in_sizes, int n_in,
                              void* d_out, int out_size, void* d_ws,
                              size_t ws_size, hipStream_t stream) {
  const float* x        = (const float*)d_in[0];
  const float* w_qkv    = (const float*)d_in[1];
  const float* reattn_w = (const float*)d_in[2];
  const float* ln_g     = (const float*)d_in[3];
  const float* ln_b     = (const float*)d_in[4];
  const float* w_out    = (const float*)d_in[5];
  const float* b_out    = (const float*)d_in[6];

  // workspace layout (bytes), all offsets 256-aligned
  char* ws = (char*)d_ws;
  const size_t XB_B    = (size_t)MTOT * DIM * 2;          // 4 MB
  const size_t WQKV_B  = (size_t)3 * INNER * DIM * 2;     // 1.5 MB
  const size_t WOUT_B  = (size_t)DIM * INNER * 2;         // 0.5 MB
  const size_t QKV1_B  = (size_t)BATCH * HEADS * NSEQ * DHEAD * 2;  // 4 MB each
  bf16* xb    = (bf16*)(ws);
  bf16* wqkvb = (bf16*)(ws + XB_B);
  bf16* woutb = (bf16*)(ws + XB_B + WQKV_B);
  bf16* qb    = (bf16*)(ws + XB_B + WQKV_B + WOUT_B);
  bf16* kb    = (bf16*)(ws + XB_B + WQKV_B + WOUT_B + QKV1_B);
  bf16* vtb   = (bf16*)(ws + XB_B + WQKV_B + WOUT_B + 2 * QKV1_B);
  bf16* aoutb = (bf16*)(ws + XB_B + WQKV_B + WOUT_B + 3 * QKV1_B);

  // 1) fp32 -> bf16
  cvt_kernel<<<(MTOT * DIM + 255) / 256, 256, 0, stream>>>(
      x, w_qkv, w_out, xb, wqkvb, woutb);

  // 2) QKV projection: 4096x1536 over K=512; 6144 wave-tiles / 8 waves
  qkv_gemm_kernel<<<768, 256, 0, stream>>>(xb, wqkvb, qb, kb, vtb);

  // 3) fused attention + re-attention: one WG per (b, 16-query tile)
  attn_kernel<<<BATCH * (NSEQ / 16), 256, 0, stream>>>(
      qb, kb, vtb, reattn_w, ln_g, ln_b, aoutb);

  // 4) output projection 4096x512 over K=512; 2048 wave-tiles / 8 waves
  out_gemm_kernel<<<256, 256, 0, stream>>>(aoutb, woutb, b_out, (float*)d_out);
}
